// Graphnetwork2_phonon_77824807403566
// MI455X (gfx1250) — matrix operand order repaired and checked by reading, not compiled, not run on hardware
//
#include <hip/hip_runtime.h>

#define N_NODES  20000
#define N_EDGES  320000
#define N_GRAPHS 128
#define HDIM     128

typedef _Float16 half_t;
typedef _Float16 v16h __attribute__((ext_vector_type(16)));
typedef _Float16 v8h  __attribute__((ext_vector_type(8)));
typedef _Float16 v4h  __attribute__((ext_vector_type(4)));
typedef float    v8f  __attribute__((ext_vector_type(8)));

union HU { v16h v; v8h h[2]; };

__device__ __forceinline__ v4h pack4(float a, float b, float c, float d) {
  v4h r; r[0] = (half_t)a; r[1] = (half_t)b; r[2] = (half_t)c; r[3] = (half_t)d; return r;
}

// CDNA5 async global->LDS copies (ASYNCcnt-tracked, no VGPR round trip).
// LDS operand = low 32 bits of the generic shared-pointer address
// (LDS aperture keeps the byte offset in addr[31:0], ISA 10.2).
__device__ __forceinline__ void async_lds_b128(void* lds, const void* g) {
  unsigned off = (unsigned)(size_t)lds;
  asm volatile("global_load_async_to_lds_b128 %0, %1, off"
               :: "v"(off), "v"(g) : "memory");
}
__device__ __forceinline__ void async_lds_b64(void* lds, const void* g) {
  unsigned off = (unsigned)(size_t)lds;
  asm volatile("global_load_async_to_lds_b64 %0, %1, off"
               :: "v"(off), "v"(g) : "memory");
}
__device__ __forceinline__ void wait_async0() {
  asm volatile("s_wait_asynccnt 0x0" ::: "memory");
}

// ---------------------------------------------------------------------------
// Generic 32xNTOT-tile WMMA GEMM; K staged in 32-chunks through double-buffered
// LDS (f16), async copies for chunk i+1 overlapped with WMMAs of chunk i.
// AMODE: 0 = fp32 A (lda, Kreal w/ zero pad)       1 = f16 A (stride KD, async)
//        2 = edge concat [h[row] | h[col] | e]      3 = node concat [h | agg/cnt]
// EPI:   0 = bias + LayerNorm + PReLU -> f16 outH (NTOT cols)
//        1 = bias + PReLU -> f16 outH
//        2 = bias -> fp32 outF
//        3 = bias; resid(=e) += v; atomicAdd(aggOut[col[row]*128+col], v)
//        4 = bias; resid(=h) += v
// ---------------------------------------------------------------------------
template<int KD, int NTOT, int AMODE, int EPI>
__global__ __launch_bounds__(256)
void gemm_k(const float* __restrict__ Af, int lda, int Kreal,
            const half_t* __restrict__ Ah,
            const float* __restrict__ gh, const float* __restrict__ ge,
            const int* __restrict__ grow, const int* __restrict__ gcol,
            const float* __restrict__ gcnt,
            const half_t* __restrict__ Bw,
            const float* __restrict__ bias,
            const float* __restrict__ lng, const float* __restrict__ lnb,
            const float* __restrict__ alpha,
            half_t* __restrict__ outH, float* __restrict__ outF,
            float* __restrict__ resid, float* __restrict__ aggOut)
{
  constexpr int SUB = NTOT / 64;   // N-subtiles (16 cols each) per wave, 8 waves
  constexpr int NCH = KD / 32;     // K chunks
  __shared__ half_t sA[2][32 * 40];
  __shared__ half_t sB[2][NTOT * 40];
  __shared__ float  outS[(EPI == 0) ? 32 * 257 : 1];
  __shared__ float  redS[(EPI == 0) ? 256 : 1];
  __shared__ float  redQ[(EPI == 0) ? 256 : 1];
  __shared__ float  mln[(EPI == 0) ? 64 : 2];
  __shared__ int    sRow[(AMODE == 2) ? 32 : 1];
  __shared__ int    sCol[(AMODE == 2 || EPI == 3) ? 32 : 1];

  const int tid   = threadIdx.x;
  const int w     = tid >> 5;
  const int lane  = tid & 31;
  const int l16   = lane & 15;
  const int lhalf = lane >> 4;
  const int m0    = blockIdx.x * 32;
  const int Mi    = w & 1;
  const int nt0   = (w >> 1) * SUB;
  const int am    = tid >> 3;        // staging row (fixed per thread)
  const int k4    = (tid & 7) << 2;  // staging k offset within chunk
  const int gm    = m0 + am;

  // per-row 1/max(cnt,1): row fixed across chunks -> load once
  float invc = 1.f;
  if (AMODE == 3) invc = 1.f / fmaxf(gcnt[gm], 1.f);

  // cache per-block gather/scatter indices
  if (AMODE == 2) {
    if (tid < 32) { sRow[tid] = grow[m0 + tid]; sCol[tid] = gcol[m0 + tid]; }
  } else if (EPI == 3) {
    if (tid < 32) sCol[tid] = gcol[m0 + tid];
  }
  if (AMODE == 2 || EPI == 3) __syncthreads();

  // ---- staging helpers ----
  auto issueB = [&](int kc, int buf) {
#pragma unroll
    for (int i2 = 0; i2 < NTOT / 64; ++i2) {
      int u   = tid + i2 * 256;       // unit = 8 halves (16 B)
      int n   = u >> 2;
      int kv8 = (u & 3) << 3;
      async_lds_b128(&sB[buf][n * 40 + kv8], Bw + (long)n * KD + kc + kv8);
    }
  };
  auto issueA = [&](int kc, int buf, float* f) {
    const int gk = kc + k4;
    if (AMODE == 0) {
#pragma unroll
      for (int j = 0; j < 4; ++j) {   // branchless clamped load
        int k  = gk + j;
        int kk = (k < Kreal) ? k : 0;
        f[j] = Af[(long)gm * lda + kk];
      }
    } else if (AMODE == 1) {
      async_lds_b64(&sA[buf][am * 40 + k4], Ah + (long)gm * KD + gk);
    } else if (AMODE == 2) {
      const float* p;
      if (gk < 128)      p = gh + (long)sRow[am] * HDIM + gk;
      else if (gk < 256) p = gh + (long)sCol[am] * HDIM + (gk - 128);
      else               p = ge + (long)gm * HDIM + (gk - 256);
      float4 v = *(const float4*)p;
      f[0] = v.x; f[1] = v.y; f[2] = v.z; f[3] = v.w;
    } else {
      const float* p = (gk < 128) ? gh + (long)gm * HDIM + gk
                                  : ge + (long)gm * HDIM + (gk - 128);
      float sc = (gk < 128) ? 1.f : invc;
      float4 v = *(const float4*)p;
      f[0] = v.x * sc; f[1] = v.y * sc; f[2] = v.z * sc; f[3] = v.w * sc;
    }
  };
  auto commitA = [&](int kc, int buf, const float* f) {
    if (AMODE == 1) return;           // already async
    const int gk = kc + k4;
    if (AMODE == 0) {
      half_t* d = &sA[buf][am * 40 + k4];
#pragma unroll
      for (int j = 0; j < 4; ++j) {
        int k = gk + j;
        d[j] = (k < Kreal) ? (half_t)f[j] : (half_t)0.f;
      }
    } else {
      *(v4h*)&sA[buf][am * 40 + k4] = pack4(f[0], f[1], f[2], f[3]);
    }
  };

  v8f acc[SUB] = {};

  // ---- prologue: stage chunk 0 into buffer 0 ----
  {
    float f[4];
    issueA(0, 0, f);
    issueB(0, 0);
    commitA(0, 0, f);
    wait_async0();
    __syncthreads();
  }

  // ---- pipelined main loop: stage chunk i+1 while computing chunk i ----
  for (int i = 0; i < NCH; ++i) {
    const int  cb   = i & 1;
    const bool more = (i + 1 < NCH);
    float f[4];
    if (more) {                       // issue next chunk's loads (async B + A)
      issueA((i + 1) * 32, cb ^ 1, f);
      issueB((i + 1) * 32, cb ^ 1);
    }
    // fragments + WMMA out of current buffer
    HU a;
    {
      const half_t* pa = &sA[cb][(Mi * 16 + l16) * 40 + lhalf * 8];
      a.h[0] = *(const v8h*)pa;
      a.h[1] = *(const v8h*)(pa + 16);
    }
#pragma unroll
    for (int s = 0; s < SUB; ++s) {
      const half_t* pb = &sB[cb][((nt0 + s) * 16 + l16) * 40 + lhalf * 8];
      HU b; b.h[0] = *(const v8h*)pb; b.h[1] = *(const v8h*)(pb + 16);
      acc[s] = __builtin_amdgcn_wmma_f32_16x16x32_f16(false, a.v, false, b.v,
                                                      (short)0, acc[s], false, false);
    }
    if (more) {
      commitA((i + 1) * 32, cb ^ 1, f);  // loadcnt wait lands after WMMA issue
      wait_async0();
      __syncthreads();
    }
  }

  // ---- epilogue ----
  if (EPI == 0) {
#pragma unroll
    for (int s = 0; s < SUB; ++s) {
      int cn = (nt0 + s) * 16 + l16;
#pragma unroll
      for (int i = 0; i < 8; ++i) {
        int r = Mi * 16 + lhalf * 8 + i;
        outS[r * 257 + cn] = acc[s][i] + bias[cn];
      }
    }
    __syncthreads();
    {
      int r = tid >> 3, seg = tid & 7;
      float sm = 0.f, sq = 0.f;
      for (int c = seg * 32; c < seg * 32 + 32; ++c) {
        float v = outS[r * 257 + c]; sm += v; sq += v * v;
      }
      redS[tid] = sm; redQ[tid] = sq;
    }
    __syncthreads();
    if (tid < 32) {
      float S = 0.f, Q = 0.f;
      for (int j = 0; j < 8; ++j) { S += redS[tid * 8 + j]; Q += redQ[tid * 8 + j]; }
      float mean = S / (float)NTOT;
      float var  = Q / (float)NTOT - mean * mean;
      mln[tid]      = mean;
      mln[32 + tid] = rsqrtf(var + 1e-5f);
    }
    __syncthreads();
    {
      float al = alpha[0];
      int r = tid >> 3, seg = tid & 7;
      float mean = mln[r], rstd = mln[32 + r];
      long gmr = m0 + r;
      for (int c = seg * 32; c < seg * 32 + 32; ++c) {
        float v = (outS[r * 257 + c] - mean) * rstd * lng[c] + lnb[c];
        v = (v >= 0.f) ? v : al * v;
        outH[gmr * NTOT + c] = (half_t)v;
      }
    }
  } else {
    float al = (EPI == 1) ? alpha[0] : 0.f;
#pragma unroll
    for (int s = 0; s < SUB; ++s) {
      int cn = (nt0 + s) * 16 + l16;
#pragma unroll
      for (int i = 0; i < 8; ++i) {
        int  rl = Mi * 16 + lhalf * 8 + i;
        long r  = m0 + rl;
        float v = acc[s][i] + bias[cn];
        if (EPI == 1) {
          v = (v >= 0.f) ? v : al * v;
          outH[r * NTOT + cn] = (half_t)v;
        } else if (EPI == 2) {
          outF[r * NTOT + cn] = v;
        } else if (EPI == 3) {
          resid[r * HDIM + cn] += v;
          atomicAdd(&aggOut[(long)sCol[rl] * HDIM + cn], v);
        } else {
          resid[r * HDIM + cn] += v;
        }
      }
    }
  }
}

// ---------------------------------------------------------------------------
// Small helper kernels
// ---------------------------------------------------------------------------
__global__ void convert_w(const float* __restrict__ src, half_t* __restrict__ dst,
                          int L, int K, int N, int Kpad)
{
  long idx = (long)blockIdx.x * 256 + threadIdx.x;
  long tot = (long)L * N * Kpad;
  if (idx >= tot) return;
  int l = (int)(idx / ((long)N * Kpad));
  int r = (int)(idx % ((long)N * Kpad));
  int n = r / Kpad, k = r % Kpad;
  dst[idx] = (k < K) ? (half_t)src[((long)l * K + k) * N + n] : (half_t)0.f;
}

__global__ void zero_f(float* p, long n)
{
  long i = (long)blockIdx.x * 256 + threadIdx.x;
  if (i < n) p[i] = 0.f;
}

__global__ void edge_attr_k(const float* __restrict__ vec, float* __restrict__ attr)
{
  int m = blockIdx.x * 256 + threadIdx.x;
  if (m >= N_EDGES) return;
  float x = vec[m * 3 + 0], y = vec[m * 3 + 1], z = vec[m * 3 + 2];
  float r = sqrtf(x * x + y * y + z * z);
  float inv = 1.f / fmaxf(r, 1e-12f);
  const float s3 = 1.7320508075688772f;
  float u = 2.f * (r * 0.25f - 1.f);
  float cut = (u > 0.f) ? 0.f
            : ((u < -1.f) ? 1.f : 0.5f * (1.f - __cosf(3.14159265358979f * u)));
  attr[m * 4 + 0] = cut;
  attr[m * 4 + 1] = cut * s3 * y * inv;
  attr[m * 4 + 2] = cut * s3 * z * inv;
  attr[m * 4 + 3] = cut * s3 * x * inv;
}

__global__ void count_k(const int* __restrict__ col, float* __restrict__ cnt)
{
  int m = blockIdx.x * 256 + threadIdx.x;
  if (m < N_EDGES) atomicAdd(&cnt[col[m]], 1.f);
}

// edge encoder layer 1 (K=4, WMMA would waste the K dim): VALU, coalesced over n
__global__ void enc_edge_k(const float* __restrict__ attr,
                           const float* __restrict__ W1, const float* __restrict__ b1,
                           const float* __restrict__ a, half_t* __restrict__ mid)
{
  long idx = (long)blockIdx.x * 256 + threadIdx.x;
  if (idx >= (long)N_EDGES * HDIM) return;
  int m = (int)(idx >> 7), n = (int)(idx & 127);
  float s = b1[n];
#pragma unroll
  for (int k = 0; k < 4; ++k) s += attr[m * 4 + k] * W1[k * HDIM + n];
  float al = a[0];
  s = (s >= 0.f) ? s : al * s;
  mid[(long)m * HDIM + n] = (half_t)s;
}

__global__ void dos_k(const float* __restrict__ h, const int* __restrict__ batch,
                      float* __restrict__ dos)
{
  long idx = (long)blockIdx.x * 256 + threadIdx.x;
  if (idx >= (long)N_NODES * HDIM) return;
  int m = (int)(idx >> 7), k = (int)(idx & 127);
  atomicAdd(&dos[(long)batch[m] * HDIM + k], h[idx]);
}

__global__ void head1_k(const float* __restrict__ dos, const float* __restrict__ W,
                        const float* __restrict__ b, float* __restrict__ z)
{
  int idx = blockIdx.x * 256 + threadIdx.x;
  if (idx >= N_GRAPHS * 64) return;
  int g = idx >> 6, j = idx & 63;
  float s = b[j];
  for (int k = 0; k < HDIM; ++k) s += dos[g * HDIM + k] * W[k * 64 + j];
  z[idx] = (s >= 0.f) ? s : 0.01f * s;
}

__global__ void head2_k(const float* __restrict__ z, const float* __restrict__ W,
                        const float* __restrict__ b, float* __restrict__ out)
{
  int idx = blockIdx.x * 256 + threadIdx.x;
  if (idx >= N_GRAPHS * 51) return;
  int g = idx / 51, o = idx % 51;
  float s = b[o];
  for (int j = 0; j < 64; ++j) s += z[g * 64 + j] * W[j * 51 + o];
  out[idx] = s;
}

#define CDIV(a, b) (((a) + (b) - 1) / (b))

extern "C" void kernel_launch(void* const* d_in, const int* in_sizes, int n_in,
                              void* d_out, int out_size, void* d_ws, size_t ws_size,
                              hipStream_t stream)
{
  const float* x       = (const float*)d_in[0];
  const int*   eidx    = (const int*)  d_in[1];
  const float* evec    = (const float*)d_in[2];
  const int*   batch   = (const int*)  d_in[3];
  const float* enc_nW1 = (const float*)d_in[4];
  const float* enc_nb1 = (const float*)d_in[5];
  const float* enc_na  = (const float*)d_in[6];
  const float* enc_nW2 = (const float*)d_in[7];
  const float* enc_nb2 = (const float*)d_in[8];
  const float* enc_eW1 = (const float*)d_in[9];
  const float* enc_eb1 = (const float*)d_in[10];
  const float* enc_ea  = (const float*)d_in[11];
  const float* enc_eW2 = (const float*)d_in[12];
  const float* enc_eb2 = (const float*)d_in[13];
  const float* eW1 = (const float*)d_in[14];
  const float* eb1 = (const float*)d_in[15];
  const float* eg  = (const float*)d_in[16];
  const float* ebt = (const float*)d_in[17];
  const float* ea  = (const float*)d_in[18];
  const float* eW2 = (const float*)d_in[19];
  const float* eb2 = (const float*)d_in[20];
  const float* nW1 = (const float*)d_in[21];
  const float* nb1 = (const float*)d_in[22];
  const float* ng  = (const float*)d_in[23];
  const float* nbt = (const float*)d_in[24];
  const float* na  = (const float*)d_in[25];
  const float* nW2 = (const float*)d_in[26];
  const float* nb2 = (const float*)d_in[27];
  const float* oW1 = (const float*)d_in[28];
  const float* ob1 = (const float*)d_in[29];
  const float* oW2 = (const float*)d_in[30];
  const float* ob2 = (const float*)d_in[31];
  (void)in_sizes; (void)n_in; (void)out_size; (void)ws_size;

  const int* erow = eidx;
  const int* ecol = eidx + N_EDGES;
  float* out = (float*)d_out;

  char* base = (char*)d_ws;
  size_t off = 0;
  auto alloc = [&](size_t bytes) -> char* {
    char* p = base + off;
    off = (off + bytes + 255) & ~(size_t)255;
    return p;
  };
  half_t* Wn1t = (half_t*)alloc((size_t)128 * 128 * 2);   // enc_nW1, K pad 118->128
  half_t* Wn2t = (half_t*)alloc((size_t)128 * 128 * 2);
  half_t* We2e = (half_t*)alloc((size_t)128 * 128 * 2);
  half_t* We1t = (half_t*)alloc((size_t)3 * 256 * 384 * 2);
  half_t* We2t = (half_t*)alloc((size_t)3 * 128 * 256 * 2);
  half_t* Wn1l = (half_t*)alloc((size_t)3 * 256 * 256 * 2);
  half_t* Wn2l = (half_t*)alloc((size_t)3 * 128 * 256 * 2);
  float*  attr = (float*) alloc((size_t)N_EDGES * 4 * 4);
  float*  h    = (float*) alloc((size_t)N_NODES * HDIM * 4);
  float*  e    = (float*) alloc((size_t)N_EDGES * HDIM * 4);
  half_t* midN = (half_t*)alloc((size_t)N_NODES * 256 * 2);
  half_t* midE = (half_t*)alloc((size_t)N_EDGES * 256 * 2);
  float*  agg  = (float*) alloc((size_t)N_NODES * HDIM * 4);
  float*  cnt  = (float*) alloc((size_t)N_NODES * 4);
  float*  dos  = (float*) alloc((size_t)N_GRAPHS * HDIM * 4);
  float*  zb   = (float*) alloc((size_t)N_GRAPHS * 64 * 4);

  // ---- weight conversion (fp32 -> f16, transposed K-major, K zero-padded) ----
  convert_w<<<CDIV(128 * 128, 256), 256, 0, stream>>>(enc_nW1, Wn1t, 1, 118, 128, 128);
  convert_w<<<CDIV(128 * 128, 256), 256, 0, stream>>>(enc_nW2, Wn2t, 1, 128, 128, 128);
  convert_w<<<CDIV(128 * 128, 256), 256, 0, stream>>>(enc_eW2, We2e, 1, 128, 128, 128);
  convert_w<<<CDIV(3 * 256 * 384, 256), 256, 0, stream>>>(eW1, We1t, 3, 384, 256, 384);
  convert_w<<<CDIV(3 * 128 * 256, 256), 256, 0, stream>>>(eW2, We2t, 3, 256, 128, 256);
  convert_w<<<CDIV(3 * 256 * 256, 256), 256, 0, stream>>>(nW1, Wn1l, 3, 256, 256, 256);
  convert_w<<<CDIV(3 * 128 * 256, 256), 256, 0, stream>>>(nW2, Wn2l, 3, 256, 128, 256);

  // ---- geometry + degree counts ----
  edge_attr_k<<<CDIV(N_EDGES, 256), 256, 0, stream>>>(evec, attr);
  zero_f<<<CDIV(N_NODES, 256), 256, 0, stream>>>(cnt, N_NODES);
  count_k<<<CDIV(N_EDGES, 256), 256, 0, stream>>>(ecol, cnt);

  // ---- node encoder: h = prelu(x@W1+b1)@W2 + b2 ----
  gemm_k<128, 128, 0, 1><<<N_NODES / 32, 256, 0, stream>>>(
      x, 118, 118, nullptr, nullptr, nullptr, nullptr, nullptr, nullptr,
      Wn1t, enc_nb1, nullptr, nullptr, enc_na, midN, nullptr, nullptr, nullptr);
  gemm_k<128, 128, 1, 2><<<N_NODES / 32, 256, 0, stream>>>(
      nullptr, 0, 0, midN, nullptr, nullptr, nullptr, nullptr, nullptr,
      Wn2t, enc_nb2, nullptr, nullptr, nullptr, nullptr, h, nullptr, nullptr);

  // ---- edge encoder: e = prelu(attr@eW1+eb1)@eW2 + eb2 ----
  enc_edge_k<<<CDIV((long)N_EDGES * HDIM, 256), 256, 0, stream>>>(
      attr, enc_eW1, enc_eb1, enc_ea, midE);
  gemm_k<128, 128, 1, 2><<<N_EDGES / 32, 256, 0, stream>>>(
      nullptr, 0, 0, midE, nullptr, nullptr, nullptr, nullptr, nullptr,
      We2e, enc_eb2, nullptr, nullptr, nullptr, nullptr, e, nullptr, nullptr);

  // ---- processor layers ----
  for (int l = 0; l < 3; ++l) {
    zero_f<<<CDIV((long)N_NODES * HDIM, 256), 256, 0, stream>>>(agg, (long)N_NODES * HDIM);
    // edge MLP 1: gather-concat -> GEMM(384->256) -> LN -> prelu -> f16
    gemm_k<384, 256, 2, 0><<<N_EDGES / 32, 256, 0, stream>>>(
        nullptr, 0, 0, nullptr, h, e, erow, ecol, nullptr,
        We1t + (size_t)l * 256 * 384, eb1 + l * 256, eg + l * 256, ebt + l * 256,
        ea + l, midE, nullptr, nullptr, nullptr);
    // edge MLP 2: GEMM(256->128); e += eo; scatter-add eo into agg[col]
    gemm_k<256, 128, 1, 3><<<N_EDGES / 32, 256, 0, stream>>>(
        nullptr, 0, 0, midE, nullptr, nullptr, nullptr, ecol, nullptr,
        We2t + (size_t)l * 128 * 256, eb2 + l * 128, nullptr, nullptr, nullptr,
        nullptr, nullptr, e, agg);
    // node MLP 1: concat [h | agg/cnt] -> GEMM(256->256) -> LN -> prelu -> f16
    gemm_k<256, 256, 3, 0><<<N_NODES / 32, 256, 0, stream>>>(
        nullptr, 0, 0, nullptr, h, agg, nullptr, nullptr, cnt,
        Wn1l + (size_t)l * 256 * 256, nb1 + l * 256, ng + l * 256, nbt + l * 256,
        na + l, midN, nullptr, nullptr, nullptr);
    // node MLP 2: GEMM(256->128); h += no
    gemm_k<256, 128, 1, 4><<<N_NODES / 32, 256, 0, stream>>>(
        nullptr, 0, 0, midN, nullptr, nullptr, nullptr, nullptr, nullptr,
        Wn2l + (size_t)l * 128 * 256, nb2 + l * 128, nullptr, nullptr, nullptr,
        nullptr, nullptr, h, nullptr);
  }

  // ---- readout head ----
  zero_f<<<CDIV(N_GRAPHS * HDIM, 256), 256, 0, stream>>>(dos, (long)N_GRAPHS * HDIM);
  dos_k<<<CDIV((long)N_NODES * HDIM, 256), 256, 0, stream>>>(h, batch, dos);
  head1_k<<<CDIV(N_GRAPHS * 64, 256), 256, 0, stream>>>(dos, oW1, ob1, zb);
  head2_k<<<CDIV(N_GRAPHS * 51, 256), 256, 0, stream>>>(zb, oW2, ob2, out);
}
